// MRF_20323785244729
// MI455X (gfx1250) — matrix operand order repaired and checked by reading, not compiled, not run on hardware
//
#include <hip/hip_runtime.h>

#define HID 256
#define NB  32
#define HH  64
#define WW  64

typedef __attribute__((ext_vector_type(16))) __bf16 v16bf;
typedef __attribute__((ext_vector_type(8)))  float  v8f;

union ABf { v16bf v; unsigned u[8]; uint4 q[2]; };

__device__ __forceinline__ unsigned short f2bf(float f) {
  unsigned u = __float_as_uint(f);
  u += 0x7FFFu + ((u >> 16) & 1u);
  return (unsigned short)(u >> 16);
}

// ---------------- prep: w_hh fp32 -> bf16 (row-major [o][k]) ----------------
__global__ void prep_whh_bf16(const float* __restrict__ w_hh,
                              unsigned short* __restrict__ whhbf) {
  int idx = blockIdx.x * blockDim.x + threadIdx.x;
  if (idx < HID * HID) whhbf[idx] = f2bf(w_hh[idx]);
}

// ---------------- row kernel: l_row = logit(i) + [k>0] * (h_below @ w_hh^T + b_hh)
// grid = 64 columns (w), 1024 threads = 32 waves = 2 n-tiles x 16 o-tiles.
__global__ void __launch_bounds__(1024)
row_kernel(const float* __restrict__ x, const float* __restrict__ w_xh,
           const float* __restrict__ b_xh, const float* __restrict__ b_hh,
           const unsigned short* __restrict__ hbf,   // h_below, bf16 (W,N,HID)
           const unsigned short* __restrict__ whhbf, // bf16 (HID,HID) row-major
           float* __restrict__ l_buf,                // f32 (W,N,HID)
           int i, int k)
{
  const int w    = blockIdx.x;
  const int lane = threadIdx.x & 31;
  const int wid  = threadIdx.x >> 5;
  const int o0   = (wid >> 1) << 4;
  const int n0   = (wid & 1) << 4;
  const bool lh  = lane < 16;
  const int oc   = o0 + (lane & 15);

  // h-term coefficient: (even w: hl valid if w>=1) / (odd w: hr valid if w<=W-2) + ht (i>=1)
  float chc = (((w & 1) == 0) ? (w >= 1 ? 1.f : 0.f) : (w <= WW - 2 ? 1.f : 0.f))
            + (i >= 1 ? 1.f : 0.f);

  const int nh[4] = { i, i, i - 1, i + 1 };
  const int nw[4] = { w - 1, w + 1, w, w };
  const float wx0 = w_xh[oc * 3 + 0];
  const float wx1 = w_xh[oc * 3 + 1];
  const float wx2 = w_xh[oc * 3 + 2];
  const float bx  = b_xh[oc];
  const float bh  = b_hh[oc];

  v8f acc;
  #pragma unroll
  for (int v = 0; v < 8; ++v) {
    int n = n0 + v + (lh ? 0 : 8);
    float s0 = 0.f, s1 = 0.f, s2 = 0.f, nbv = 0.f;
    #pragma unroll
    for (int t = 0; t < 4; ++t) {
      int hh = nh[t], ww = nw[t];
      if (hh >= 0 && hh < HH && ww >= 0 && ww < WW) {
        const float* xp = x + (((size_t)n * 3) * HH + hh) * WW + ww;
        s0 += xp[0];
        s1 += xp[(size_t)HH * WW];
        s2 += xp[2 * (size_t)HH * WW];
        nbv += 1.f;
      }
    }
    float l0 = chc * bh + s0 * wx0 + s1 * wx1 + s2 * wx2 + nbv * bx;
    acc[v] = l0 + (k > 0 ? bh : 0.f);
  }

  if (k > 0) {
    const int m = n0 + (lane & 15);
    const unsigned short* arow = hbf + ((size_t)(w * NB + m)) * HID;
    const unsigned short* brow = whhbf + (size_t)oc * HID;
    const int abase = lh ? 0 : 8;
    const int bbase = lh ? 0 : 8;
    #pragma unroll
    for (int kk = 0; kk < 8; ++kk) {
      const int kb = kk * 32;
      ABf A, B;
      A.q[0] = *(const uint4*)(arow + kb + abase);
      A.q[1] = *(const uint4*)(arow + kb + 16 + abase);
      B.q[0] = *(const uint4*)(brow + kb + bbase);
      B.q[1] = *(const uint4*)(brow + kb + bbase + 16);
      acc = __builtin_amdgcn_wmma_f32_16x16x32_bf16(false, A.v, false, B.v,
                                                    (short)0, acc, false, false);
    }
  }

  #pragma unroll
  for (int v = 0; v < 8; ++v) {
    int n = n0 + v + (lh ? 0 : 8);
    l_buf[((size_t)(w * NB + n)) * HID + oc] = acc[v];
  }
}

// ---------------- column-scan kernel: one workgroup runs all 64 serpentine steps.
// B tile (16 o-channels x 256 K, bf16) held in 64 VGPRs for the whole kernel;
// h_prev carried in padded LDS (row stride 264 -> conflict-free b128 reads).
__global__ void __launch_bounds__(1024)
col_kernel(const float* __restrict__ l_buf, const float* __restrict__ b_hh,
           const unsigned short* __restrict__ whhbf,
           unsigned short* __restrict__ hbf,     // h_row out, bf16 (W,N,HID)
           float* __restrict__ out, int i)
{
  __shared__ __attribute__((aligned(16))) unsigned short s_h[NB * 264];

  const int lane = threadIdx.x & 31;
  const int wid  = threadIdx.x >> 5;
  const int o0   = (wid >> 1) << 4;
  const int n0   = (wid & 1) << 4;
  const bool lh  = lane < 16;
  const int oc   = o0 + (lane & 15);
  const int m    = n0 + (lane & 15);
  const float bh = b_hh[oc];
  const bool flip = ((i & 1) == 0);
  const int abase = lh ? 0 : 8;
  const int bbase = lh ? 0 : 8;

  // B tile into registers once (invariant over all 64 column steps)
  ABf Breg[8];
  {
    const unsigned short* brow = whhbf + (size_t)oc * HID;
    #pragma unroll
    for (int kk = 0; kk < 8; ++kk) {
      Breg[kk].q[0] = *(const uint4*)(brow + kk * 32 + bbase);
      Breg[kk].q[1] = *(const uint4*)(brow + kk * 32 + bbase + 16);
    }
  }

  for (int j = 0; j < WW; ++j) {
    const int cw = flip ? (WW - 1 - j) : j;
    const float* lrow = l_buf + (size_t)cw * NB * HID;

    v8f acc;
    #pragma unroll
    for (int v = 0; v < 8; ++v) {
      int n = n0 + v + (lh ? 0 : 8);
      acc[v] = lrow[(size_t)n * HID + oc];
    }

    if (j > 0) {
      #pragma unroll
      for (int v = 0; v < 8; ++v) acc[v] += bh;
      const unsigned short* arow = s_h + (size_t)m * 264;
      #pragma unroll
      for (int kk = 0; kk < 8; ++kk) {
        const int kb = kk * 32;
        ABf A;
        A.q[0] = *(const uint4*)(arow + kb + abase);
        A.q[1] = *(const uint4*)(arow + kb + 16 + abase);
        acc = __builtin_amdgcn_wmma_f32_16x16x32_bf16(false, A.v, false, Breg[kk].v,
                                                      (short)0, acc, false, false);
      }
    }

    float hval[8];
    #pragma unroll
    for (int v = 0; v < 8; ++v) hval[v] = 1.f / (1.f + __expf(-acc[v]));

    __syncthreads();  // all waves finished reading s_h for this step
    #pragma unroll
    for (int v = 0; v < 8; ++v) {
      int n = n0 + v + (lh ? 0 : 8);
      unsigned short hb = f2bf(hval[v]);
      s_h[n * 264 + oc] = hb;                                    // h_prev for next step
      hbf[((size_t)(cw * NB + n)) * HID + oc] = hb;              // h_below for next row
      out[(((size_t)n * HID + oc) * HH + i) * WW + cw] = hval[v];// final output (N,HID,H,W)
    }
    __syncthreads();  // s_h published for step j+1
  }
}

extern "C" void kernel_launch(void* const* d_in, const int* in_sizes, int n_in,
                              void* d_out, int out_size, void* d_ws, size_t ws_size,
                              hipStream_t stream)
{
  const float* x    = (const float*)d_in[0];
  const float* w_xh = (const float*)d_in[1];
  const float* b_xh = (const float*)d_in[2];
  const float* w_hh = (const float*)d_in[3];
  const float* b_hh = (const float*)d_in[4];
  float* out = (float*)d_out;

  char* ws = (char*)d_ws;
  unsigned short* whhbf = (unsigned short*)ws;                       // 131072 B
  unsigned short* hbf   = (unsigned short*)(ws + 131072);            // 1 MiB
  float*          l_buf = (float*)(ws + 131072 + 1048576);           // 2 MiB

  prep_whh_bf16<<<256, 256, 0, stream>>>(w_hh, whhbf);

  for (int k = 0; k < HH; ++k) {
    int i = HH - 1 - k;
    row_kernel<<<WW, 1024, 0, stream>>>(x, w_xh, b_xh, b_hh, hbf, whhbf, l_buf, i, k);
    col_kernel<<<1, 1024, 0, stream>>>(l_buf, b_hh, whhbf, hbf, out, i);
  }
}
